// PairWiseCrossAttention_10007273800217
// MI455X (gfx1250) — compile-verified, matrix operand order checked by reading, not run on hardware
//
#include <hip/hip_runtime.h>

// ---------------------------------------------------------------------------
// Types for CDNA5 WMMA (wave32, 16x16x32 bf16 -> f32)
// ---------------------------------------------------------------------------
typedef __bf16 bf16_t;
typedef __attribute__((ext_vector_type(16))) __bf16 v16bf;
typedef __attribute__((ext_vector_type(8)))  __bf16 v8bf;
typedef __attribute__((ext_vector_type(4)))  __bf16 v4bf;
typedef __attribute__((ext_vector_type(8)))  float  v8f;
typedef __attribute__((ext_vector_type(4)))  int    v4i_t;

// Async global->LDS staging (gfx1250 GLOBAL_LOAD_ASYNC_TO_LDS_B128, ASYNCcnt).
#if defined(__AMDGCN__) && \
    __has_builtin(__builtin_amdgcn_global_load_async_to_lds_b128) && \
    __has_builtin(__builtin_amdgcn_s_wait_asynccnt)
#define HAVE_ASYNC_LDS 1
#else
#define HAVE_ASYNC_LDS 0
#endif

#if HAVE_ASYNC_LDS
typedef v4i_t __attribute__((address_space(1)))* as1_v4i;
typedef v4i_t __attribute__((address_space(3)))* as3_v4i;
template <int OFF>
__device__ __forceinline__ void async_b128(const void* g, void* l) {
    __builtin_amdgcn_global_load_async_to_lds_b128(
        (as1_v4i)const_cast<void*>(g), (as3_v4i)l, OFF, 0);
}
#else
template <int OFF>
__device__ __forceinline__ void async_b128(const void*, void*) {}
#endif

union FragU { v16bf v; v8bf h[2]; };

// Per-lane 16-bit A/B fragment for one 16x32 tile, per ISA layout:
// lane L(0..15)=row, khalf=lane>>4 selects K-offset 8*khalf; values are two
// contiguous 8-element runs: [8*khalf, +8) and [16+8*khalf, +8).
__device__ __forceinline__ v16bf ld_frag32(const bf16_t* rowp, int khalf) {
    FragU f;
    f.h[0] = *reinterpret_cast<const v8bf*>(rowp + khalf * 8);
    f.h[1] = *reinterpret_cast<const v8bf*>(rowp + 16 + khalf * 8);
    return f.v;
}

__device__ __forceinline__ v8f wmma_bf16(v16bf a, v16bf b, v8f c) {
    return __builtin_amdgcn_wmma_f32_16x16x32_bf16(
        /*neg_a=*/false, a, /*neg_b=*/false, b,
        /*c_mod=*/(short)0, c, /*reuse_a=*/false, /*reuse_b=*/false);
}

// ---------------------------------------------------------------------------
// Problem constants
// ---------------------------------------------------------------------------
#define BDIM 8
#define NSEQ 1024
#define DDIM 768
#define HDIM 12
#define HD   64
#define MROWS (BDIM * NSEQ)       // 8192
#define KV2N  (2 * NSEQ)          // 2048

// ---------------------------------------------------------------------------
// f32 -> bf16 conversion (4-wide)
// ---------------------------------------------------------------------------
__global__ void cvt_f32_bf16(const float* __restrict__ s, bf16_t* __restrict__ d, int n4) {
    int i = blockIdx.x * blockDim.x + threadIdx.x;
    if (i < n4) {
        float4 f = reinterpret_cast<const float4*>(s)[i];
        v4bf o;
        o[0] = (bf16_t)f.x; o[1] = (bf16_t)f.y; o[2] = (bf16_t)f.z; o[3] = (bf16_t)f.w;
        reinterpret_cast<v4bf*>(d)[i] = o;
    }
}

// ---------------------------------------------------------------------------
// Tiled bf16 GEMM:  out[m][n] = sum_k A[m][k] * W[n][k]    (i.e. A @ W^T)
// M=8192, N=768, K=768.  BM=BN=128, BK=32.  256 threads = 8 waves (2x4 grid),
// each wave owns a 64x32 region = 4x2 WMMA tiles.
// mode 0: write bf16 into head-major (B,H,seqTotal,HD) at seq offset seqOff
// mode 1: write f32 + bias into plain (M,N) layout (final Wo projection)
// ---------------------------------------------------------------------------
__global__ __launch_bounds__(256) void gemm_bf16_kernel(
    const bf16_t* __restrict__ A, const bf16_t* __restrict__ W,
    bf16_t* __restrict__ dstb, float* __restrict__ dstf,
    const float* __restrict__ bias, int mode, int seqTotal, int seqOff)
{
    __shared__ __align__(16) bf16_t As[128][32];
    __shared__ __align__(16) bf16_t Bs[128][32];

    const int tid   = threadIdx.x;
    const int lane  = tid & 31;
    const int wv    = tid >> 5;
    const int wm    = wv >> 2;          // 0..1
    const int wn    = wv & 3;           // 0..3
    const int khalf = lane >> 4;
    const int l16   = lane & 15;

    const int n0 = blockIdx.x * 128;
    const int m0 = blockIdx.y * 128;

    v8f acc[4][2];
#pragma unroll
    for (int i = 0; i < 4; ++i)
#pragma unroll
        for (int j = 0; j < 2; ++j)
#pragma unroll
            for (int r = 0; r < 8; ++r) acc[i][j][r] = 0.0f;

    const int srow = tid >> 1;          // 0..127
    const int sseg = tid & 1;           // 16-half segment
    const bf16_t* aptr = A + (size_t)(m0 + srow) * DDIM + sseg * 16;
    const bf16_t* wptr = W + (size_t)(n0 + srow) * DDIM + sseg * 16;
    bf16_t* lA = &As[srow][sseg * 16];
    bf16_t* lB = &Bs[srow][sseg * 16];

    for (int kt = 0; kt < DDIM / 32; ++kt) {
        __syncthreads();
        {
#if HAVE_ASYNC_LDS
            async_b128<0>(aptr + kt * 32, lA);
            async_b128<16>(aptr + kt * 32, lA);
            async_b128<0>(wptr + kt * 32, lB);
            async_b128<16>(wptr + kt * 32, lB);
#else
            const uint4* ga = reinterpret_cast<const uint4*>(aptr + kt * 32);
            const uint4* gw = reinterpret_cast<const uint4*>(wptr + kt * 32);
            uint4 a0 = ga[0], a1 = ga[1];
            uint4 w0 = gw[0], w1 = gw[1];
            reinterpret_cast<uint4*>(lA)[0] = a0;
            reinterpret_cast<uint4*>(lA)[1] = a1;
            reinterpret_cast<uint4*>(lB)[0] = w0;
            reinterpret_cast<uint4*>(lB)[1] = w1;
#endif
            if (kt + 1 < DDIM / 32) {   // gfx1250 global_prefetch_b8
                __builtin_prefetch(aptr + (kt + 1) * 32, 0, 1);
                __builtin_prefetch(wptr + (kt + 1) * 32, 0, 1);
            }
        }
#if HAVE_ASYNC_LDS
        __builtin_amdgcn_s_wait_asynccnt(0);
#endif
        __syncthreads();

        v16bf af[4], bfr[2];
#pragma unroll
        for (int mt = 0; mt < 4; ++mt)
            af[mt] = ld_frag32(&As[wm * 64 + mt * 16 + l16][0], khalf);
#pragma unroll
        for (int nt = 0; nt < 2; ++nt)
            bfr[nt] = ld_frag32(&Bs[wn * 32 + nt * 16 + l16][0], khalf);
#pragma unroll
        for (int mt = 0; mt < 4; ++mt)
#pragma unroll
            for (int nt = 0; nt < 2; ++nt)
                acc[mt][nt] = wmma_bf16(af[mt], bfr[nt], acc[mt][nt]);
    }

    // Epilogue. C-layout: VGPR r holds (M = r + 8*(lane>=16), N = lane&15).
    const int mofs = (lane >> 4) << 3;
#pragma unroll
    for (int mt = 0; mt < 4; ++mt) {
#pragma unroll
        for (int nt = 0; nt < 2; ++nt) {
#pragma unroll
            for (int r = 0; r < 8; ++r) {
                const int m = m0 + wm * 64 + mt * 16 + r + mofs;
                const int n = n0 + wn * 32 + nt * 16 + l16;
                const float v = acc[mt][nt][r];
                if (mode == 0) {
                    const int bidx = m >> 10, nseq = m & (NSEQ - 1);
                    const int h = n >> 6, hd = n & (HD - 1);
                    const size_t dst =
                        (((size_t)(bidx * HDIM + h)) * seqTotal + nseq + seqOff) * HD + hd;
                    dstb[dst] = (bf16_t)v;
                } else {
                    dstf[(size_t)m * DDIM + n] = v + bias[n];
                }
            }
        }
    }
}

// ---------------------------------------------------------------------------
// Flash attention: one block per (b*H+h, qtile of 64 rows). 128 threads =
// 4 waves, each wave owns 16 query rows. 64-key chunks; online softmax.
// ---------------------------------------------------------------------------
__global__ __launch_bounds__(128) void attn_kernel(
    const bf16_t* __restrict__ q,   // (B,H,N,HD)
    const bf16_t* __restrict__ kc,  // (B,H,2N,HD)
    const bf16_t* __restrict__ vc,  // (B,H,2N,HD)
    bf16_t* __restrict__ attO)      // (B,N,D)
{
    __shared__ __align__(16) bf16_t Qs[64][64];
    __shared__ __align__(16) bf16_t Ks[64][64];
    __shared__ __align__(16) bf16_t Vts[64][64];   // transposed: Vts[hd][key]
    __shared__ __align__(16) bf16_t Ps[4][16][64]; // per-wave P tile

    const int bh  = blockIdx.x;          // 0..95
    const int qt  = blockIdx.y;          // 0..15
    const int tid = threadIdx.x;
    const int lane = tid & 31;
    const int w    = tid >> 5;
    const int khalf = lane >> 4;
    const int l16   = lane & 15;
    const int mofs  = khalf << 3;

    const int row = tid >> 1, seg = tid & 1;   // staging coordinates

    // --- load Q tile once (64 rows x 64 halves) ---
    {
        const bf16_t* src = q + (((size_t)bh * NSEQ + qt * 64 + row) * HD) + seg * 32;
        bf16_t* dst = &Qs[row][seg * 32];
#if HAVE_ASYNC_LDS
        async_b128<0>(src, dst);
        async_b128<16>(src, dst);
        async_b128<32>(src, dst);
        async_b128<48>(src, dst);
#else
#pragma unroll
        for (int i = 0; i < 4; ++i)
            reinterpret_cast<v8bf*>(dst)[i] = reinterpret_cast<const v8bf*>(src)[i];
#endif
    }

    float m_i[8], l_i[8];
    v8f oacc[4];
#pragma unroll
    for (int r = 0; r < 8; ++r) { m_i[r] = -1e30f; l_i[r] = 0.0f; }
#pragma unroll
    for (int nt = 0; nt < 4; ++nt)
#pragma unroll
        for (int r = 0; r < 8; ++r) oacc[nt][r] = 0.0f;

    const float scale = 0.125f;  // 1/sqrt(64)

    for (int c = 0; c < KV2N / 64; ++c) {
        __syncthreads();
        // stage K chunk (row-major, async) and V chunk (transposed, sync)
        {
            const bf16_t* ksrc =
                kc + (((size_t)bh * KV2N + c * 64 + row) * HD) + seg * 32;
            bf16_t* kdst = &Ks[row][seg * 32];
#if HAVE_ASYNC_LDS
            async_b128<0>(ksrc, kdst);
            async_b128<16>(ksrc, kdst);
            async_b128<32>(ksrc, kdst);
            async_b128<48>(ksrc, kdst);
#else
#pragma unroll
            for (int i = 0; i < 4; ++i)
                reinterpret_cast<v8bf*>(kdst)[i] =
                    reinterpret_cast<const v8bf*>(ksrc)[i];
#endif
            const int key = row, hd0 = seg * 32;
            const v8bf* vsrc = reinterpret_cast<const v8bf*>(
                vc + (((size_t)bh * KV2N + c * 64 + key) * HD) + hd0);
#pragma unroll
            for (int i = 0; i < 4; ++i) {
                v8bf vv = vsrc[i];
#pragma unroll
                for (int j = 0; j < 8; ++j) Vts[hd0 + i * 8 + j][key] = vv[j];
            }
        }
#if HAVE_ASYNC_LDS
        __builtin_amdgcn_s_wait_asynccnt(0);
#endif
        __syncthreads();

        // ---- S = Q K^T  (16 rows x 64 keys per wave) ----
        v8f s[4];
#pragma unroll
        for (int nt = 0; nt < 4; ++nt)
#pragma unroll
            for (int r = 0; r < 8; ++r) s[nt][r] = 0.0f;
#pragma unroll
        for (int ks = 0; ks < 2; ++ks) {
            v16bf aq = ld_frag32(&Qs[w * 16 + l16][ks * 32], khalf);
#pragma unroll
            for (int nt = 0; nt < 4; ++nt) {
                v16bf bk = ld_frag32(&Ks[nt * 16 + l16][ks * 32], khalf);
                s[nt] = wmma_bf16(aq, bk, s[nt]);
            }
        }
        // scale logits
#pragma unroll
        for (int nt = 0; nt < 4; ++nt)
#pragma unroll
            for (int r = 0; r < 8; ++r) s[nt][r] *= scale;

        // ---- online softmax ----
        float mnew[8], alpha[8], rs[8];
#pragma unroll
        for (int r = 0; r < 8; ++r) {
            float v = s[0][r];
            v = fmaxf(v, s[1][r]); v = fmaxf(v, s[2][r]); v = fmaxf(v, s[3][r]);
            v = fmaxf(v, __shfl_xor(v, 1, 32));
            v = fmaxf(v, __shfl_xor(v, 2, 32));
            v = fmaxf(v, __shfl_xor(v, 4, 32));
            v = fmaxf(v, __shfl_xor(v, 8, 32));      // stays in 16-lane halves
            mnew[r]  = fmaxf(m_i[r], v);
            alpha[r] = __expf(m_i[r] - mnew[r]);
            rs[r]    = 0.0f;
        }
#pragma unroll
        for (int nt = 0; nt < 4; ++nt) {
#pragma unroll
            for (int r = 0; r < 8; ++r) {
                float p = __expf(s[nt][r] - mnew[r]);
                rs[r] += p;
                Ps[w][r + mofs][nt * 16 + l16] = (bf16_t)p;
            }
        }
#pragma unroll
        for (int r = 0; r < 8; ++r) {
            float t = rs[r];
            t += __shfl_xor(t, 1, 32);
            t += __shfl_xor(t, 2, 32);
            t += __shfl_xor(t, 4, 32);
            t += __shfl_xor(t, 8, 32);
            l_i[r] = l_i[r] * alpha[r] + t;
            m_i[r] = mnew[r];
        }
#pragma unroll
        for (int nt = 0; nt < 4; ++nt)
#pragma unroll
            for (int r = 0; r < 8; ++r) oacc[nt][r] *= alpha[r];

        // ---- O += P V  (A = P from LDS, B = V^T rows -> contiguous) ----
#pragma unroll
        for (int ks = 0; ks < 2; ++ks) {
            v16bf ap = ld_frag32(&Ps[w][l16][ks * 32], khalf);
#pragma unroll
            for (int nt = 0; nt < 4; ++nt) {
                v16bf bv = ld_frag32(&Vts[nt * 16 + l16][ks * 32], khalf);
                oacc[nt] = wmma_bf16(ap, bv, oacc[nt]);
            }
        }
    }

    // ---- finalize: O /= l, write bf16 to (B,N,D) for the Wo GEMM ----
    const int b = bh / HDIM, h = bh % HDIM;
#pragma unroll
    for (int nt = 0; nt < 4; ++nt) {
#pragma unroll
        for (int r = 0; r < 8; ++r) {
            const int nseq = qt * 64 + w * 16 + r + mofs;
            const int col  = h * HD + nt * 16 + l16;
            const float o  = oacc[nt][r] * (1.0f / l_i[r]);
            attO[((size_t)b * NSEQ + nseq) * DDIM + col] = (bf16_t)o;
        }
    }
}

// ---------------------------------------------------------------------------
// Launch
// ---------------------------------------------------------------------------
extern "C" void kernel_launch(void* const* d_in, const int* in_sizes, int n_in,
                              void* d_out, int out_size, void* d_ws, size_t ws_size,
                              hipStream_t stream) {
    const float* x1 = (const float*)d_in[0];
    const float* x2 = (const float*)d_in[1];
    const float* Wq = (const float*)d_in[2];
    const float* Wk = (const float*)d_in[3];
    const float* Wv = (const float*)d_in[4];
    const float* Wo = (const float*)d_in[5];
    const float* bo = (const float*)d_in[6];
    float* out = (float*)d_out;

    const size_t SZ_X = (size_t)MROWS * DDIM;          // 6,291,456
    const size_t SZ_W = (size_t)DDIM * DDIM;           //   589,824
    const size_t SZ_Q = (size_t)BDIM * HDIM * NSEQ * HD;
    const size_t SZ_K = (size_t)BDIM * HDIM * KV2N * HD;

    size_t off = 0;
    auto alloc = [&](size_t elems) {
        bf16_t* p = (bf16_t*)((char*)d_ws + off);
        off += elems * sizeof(bf16_t);
        return p;
    };
    bf16_t* x1b = alloc(SZ_X);
    bf16_t* x2b = alloc(SZ_X);
    bf16_t* Wqb = alloc(SZ_W);
    bf16_t* Wkb = alloc(SZ_W);
    bf16_t* Wvb = alloc(SZ_W);
    bf16_t* Wob = alloc(SZ_W);
    bf16_t* qb  = alloc(SZ_Q);
    bf16_t* kcb = alloc(SZ_K);
    bf16_t* vcb = alloc(SZ_K);
    bf16_t* aOb = alloc(SZ_X);

    auto cvt = [&](const float* s, bf16_t* d, size_t n) {
        int n4 = (int)(n / 4);
        cvt_f32_bf16<<<(n4 + 255) / 256, 256, 0, stream>>>(s, d, n4);
    };
    cvt(x1, x1b, SZ_X);
    cvt(x2, x2b, SZ_X);
    cvt(Wq, Wqb, SZ_W);
    cvt(Wk, Wkb, SZ_W);
    cvt(Wv, Wvb, SZ_W);
    cvt(Wo, Wob, SZ_W);

    dim3 ggrid(DDIM / 128, MROWS / 128);   // (6, 64)
    // q1 = x1 @ Wq^T -> (B,H,N,HD)
    gemm_bf16_kernel<<<ggrid, 256, 0, stream>>>(x1b, Wqb, qb,  nullptr, nullptr, 0, NSEQ, 0);
    // k1,k2 -> kc (B,H,2N,HD);  v1,v2 -> vc
    gemm_bf16_kernel<<<ggrid, 256, 0, stream>>>(x1b, Wkb, kcb, nullptr, nullptr, 0, KV2N, 0);
    gemm_bf16_kernel<<<ggrid, 256, 0, stream>>>(x2b, Wkb, kcb, nullptr, nullptr, 0, KV2N, NSEQ);
    gemm_bf16_kernel<<<ggrid, 256, 0, stream>>>(x1b, Wvb, vcb, nullptr, nullptr, 0, KV2N, 0);
    gemm_bf16_kernel<<<ggrid, 256, 0, stream>>>(x2b, Wvb, vcb, nullptr, nullptr, 0, KV2N, NSEQ);

    dim3 agrid(BDIM * HDIM, NSEQ / 64);    // (96, 16)
    attn_kernel<<<agrid, 128, 0, stream>>>(qb, kcb, vcb, aOb);

    // out = attO @ Wo^T + bo  (f32 output)
    gemm_bf16_kernel<<<ggrid, 256, 0, stream>>>(aOb, Wob, nullptr, out, bo, 1, 0, 0);
}
